// GraphRec_72945724555841
// MI455X (gfx1250) — matrix operand order repaired and checked by reading, not compiled, not run on hardware
//
#include <hip/hip_runtime.h>
#include <hip/hip_bf16.h>

// GraphRec fused kernel for MI455X (gfx1250, wave32).
//
// Attention logits collapse to dot products with precomputed vectors
// u = Wq@Wa, v = Wk@Wa, w = Wa (since logits are immediately projected by the
// Dx1 matrix Wa), eliminating the [B*L,64]x[64,64] GEMMs. The kernel is then
// L2-resident-gather bound (embedding tables: 2 x 25.6MB << 192MB L2).
// The remaining GEMM (fc1: [B,128]@[128,64]) runs on v_wmma_f32_16x16x32_f16,
// one 16-row tile per workgroup, B-matrix pre-packed into the wave32 WMMA
// fragment layout so each fragment is two coalesced b128 loads.

typedef __attribute__((ext_vector_type(16))) _Float16 v16h;
typedef __attribute__((ext_vector_type(8)))  _Float16 v8h;
typedef __attribute__((ext_vector_type(8)))  float    v8f;

#define D_DIM     64
#define LMAX      50
#define WAVES     4      // waves per block (128 threads)
#define BPW       4      // batch elements per wave -> 16 per block
#define ROWSTRIDE 66     // floats: conflict-free LDS row stride for neighbor stage
#define CSTRIDE   136    // halves: conflict-free LDS row stride for concat tile
#define PACK_OFF  2048   // byte offset of packed fc1_w inside d_ws

__device__ __forceinline__ float wave_sum(float x) {
  #pragma unroll
  for (int m = 16; m; m >>= 1) x += __shfl_xor(x, m, 32);
  return x;
}
__device__ __forceinline__ float wave_max(float x) {
  #pragma unroll
  for (int m = 16; m; m >>= 1) x = fmaxf(x, __shfl_xor(x, m, 32));
  return x;
}

// One attention block, executed by one wave for one batch element.
//   score[l] = q.u + e_l.v + (HASR ? r_l * (e_l.w) : 0);  softmax over l;
//   return sum_l softmax_l * e_l   (lane owns dims 2*lane, 2*lane+1)
template<bool HASR>
__device__ __forceinline__ float2 wave_attention(
    const float* __restrict__ table,
    const int*   __restrict__ idxs,
    const float* __restrict__ rats,
    int L, float2 q2,
    const float* __restrict__ uvec,   // global: Wq@Wa
    const float* __restrict__ svw,    // LDS: [0:64)=v (Wk@Wa), [64:128)=w (Wa)
    float* __restrict__ est,          // LDS: this wave's L x ROWSTRIDE stage
    int lane)
{
  // q . u  (wave reduction)
  float2 uu = *(const float2*)(uvec + 2 * lane);
  float qdotu = wave_sum(q2.x * uu.x + q2.y * uu.y);

  // Stage neighbor embeddings (coalesced float2 per lane per row).
  for (int l = 0; l < L; ++l) {
    int it = idxs[l];
    float2 e = *(const float2*)(table + (size_t)it * D_DIM + 2 * lane);
    *(float2*)(est + l * ROWSTRIDE + 2 * lane) = e;
  }
  __builtin_amdgcn_wave_barrier();  // keep stage stores before cross-lane reads

  // Scores: lane handles rows l0=lane, l1=lane+32 (bank-conflict-free reads).
  const int l0 = lane, l1 = lane + 32;
  const bool v1ok = (l1 < L);
  float dv0 = 0.f, dw0 = 0.f, dv1 = 0.f, dw1 = 0.f;
  {
    const float* rp = est + l0 * ROWSTRIDE;
    for (int d = 0; d < D_DIM; ++d) {
      float e = rp[d];
      dv0 += e * svw[d];
      if constexpr (HASR) dw0 += e * svw[D_DIM + d];
    }
  }
  if (v1ok) {
    const float* rp = est + l1 * ROWSTRIDE;
    for (int d = 0; d < D_DIM; ++d) {
      float e = rp[d];
      dv1 += e * svw[d];
      if constexpr (HASR) dw1 += e * svw[D_DIM + d];
    }
  }
  float s0 = qdotu + dv0;
  float s1 = qdotu + dv1;
  if constexpr (HASR) {
    s0 += rats[l0] * dw0;
    if (v1ok) s1 += rats[l1] * dw1;
  }
  if (!v1ok) s1 = -3.0e38f;

  // Softmax over L (wave reductions).
  float m  = wave_max(fmaxf(s0, s1));
  float e0 = __expf(s0 - m);
  float e1 = v1ok ? __expf(s1 - m) : 0.0f;
  float inv = 1.0f / wave_sum(e0 + e1);
  float w0 = e0 * inv, w1 = e1 * inv;

  // Weighted sum of neighbors (lane owns two dims).
  float2 acc; acc.x = 0.f; acc.y = 0.f;
  for (int l = 0; l < L; ++l) {
    float wl = (l < 32) ? __shfl(w0, l, 32) : __shfl(w1, l - 32, 32);
    float2 e = *(const float2*)(est + l * ROWSTRIDE + 2 * lane);
    acc.x += wl * e.x;
    acc.y += wl * e.y;
  }
  return acc;
}

// Precompute the six projection vectors: u_k = Wq_k @ Wa_k, v_k = Wk_k @ Wa_k.
__global__ void graphrec_prep(const float* __restrict__ W1, const float* __restrict__ W2,
                              const float* __restrict__ W3, const float* __restrict__ W4,
                              const float* __restrict__ W5, const float* __restrict__ W6,
                              const float* __restrict__ W7, const float* __restrict__ W8,
                              const float* __restrict__ W9, float* __restrict__ proj)
{
  int t = threadIdx.x;
  if (t >= 6 * D_DIM) return;
  int g = t >> 6, i = t & 63;
  const float *Wm, *Wa;
  switch (g) {
    case 0:  Wm = W1; Wa = W3; break;  // u1
    case 1:  Wm = W2; Wa = W3; break;  // v1
    case 2:  Wm = W4; Wa = W6; break;  // u2
    case 3:  Wm = W5; Wa = W6; break;  // v2
    case 4:  Wm = W7; Wa = W9; break;  // u3
    default: Wm = W8; Wa = W9; break;  // v3
  }
  float s = 0.f;
  for (int j = 0; j < D_DIM; ++j) s += Wm[i * D_DIM + j] * Wa[j];
  proj[t] = s;
}

// Pre-pack fc1_w (f32 [128,64]) into f16 wave32 WMMA B-fragments:
//   pack[(((nn*4 + kk)*32 + lane)*16) + j]
// with h = lane/16: element j -> (K = kk*32 + 8h + j (j<8) or kk*32 + 16 + 8h + (j-8) (j>=8),
//                                N = nn*16 + lane%16)
__global__ void graphrec_pack_b(const float* __restrict__ fc1_w,
                                _Float16* __restrict__ pack)
{
  int t = blockIdx.x * blockDim.x + threadIdx.x;   // 0 .. 8191
  if (t >= 4 * 4 * 32 * 16) return;
  int j    = t & 15;
  int lane = (t >> 4) & 31;
  int kk   = (t >> 9) & 3;
  int nn   = t >> 11;
  int K = kk * 32 + (j & 7) + ((j >> 3) << 4) + ((lane >> 4) << 3);
  int N = nn * 16 + (lane & 15);
  pack[t] = (_Float16)fc1_w[(size_t)K * D_DIM + N];
}

__global__ __launch_bounds__(WAVES * 32)
void graphrec_main(const int* __restrict__ user_ids, const int* __restrict__ item_ids,
                   const int* __restrict__ uh_items, const float* __restrict__ uh_rat,
                   const int* __restrict__ u_friends,
                   const int* __restrict__ ih_users, const float* __restrict__ ih_rat,
                   const float* __restrict__ embed_user, const float* __restrict__ embed_item,
                   const float* __restrict__ W3, const float* __restrict__ W6,
                   const float* __restrict__ W9,
                   const _Float16* __restrict__ fc1_pack, const float* __restrict__ fc1_b,
                   const float* __restrict__ fc2_w, const float* __restrict__ fc2_b,
                   const float* __restrict__ proj, float* __restrict__ out)
{
  __shared__ float    s_e[WAVES][LMAX * ROWSTRIDE]; // neighbor stage per wave
  __shared__ float    s_vw[2 * D_DIM];              // v then w, per attention phase
  __shared__ _Float16 s_cat[16 * CSTRIDE];          // concat tile (16 rows x 128, padded)
  __shared__ float    s_red[WAVES * 16];            // per-wave partial row sums

  const int tid  = threadIdx.x;
  const int lane = tid & 31;
  const int wave = tid >> 5;
  const int b0   = blockIdx.x * (WAVES * BPW);
  float* est = s_e[wave];

  float2 qu[BPW], qi[BPW], au[BPW], ai[BPW];
  #pragma unroll
  for (int ib = 0; ib < BPW; ++ib) {
    int b = b0 + wave * BPW + ib;
    int uid = user_ids[b], iid = item_ids[b];
    qu[ib] = *(const float2*)(embed_user + (size_t)uid * D_DIM + 2 * lane);
    qi[ib] = *(const float2*)(embed_item + (size_t)iid * D_DIM + 2 * lane);
    au[ib] = qu[ib];
    ai[ib] = qi[ib];
  }

  // ---- attention 1: user <- rated items (W1,W2,W3) ----
  if (tid < D_DIM) s_vw[tid] = proj[64 + tid];        // v1 = W2@W3
  else             s_vw[tid] = W3[tid - D_DIM];       // w1 = W3
  __syncthreads();
  for (int ib = 0; ib < BPW; ++ib) {
    int b = b0 + wave * BPW + ib;
    float2 o = wave_attention<true>(embed_item, uh_items + (size_t)b * 50,
                                    uh_rat + (size_t)b * 50, 50, qu[ib],
                                    proj + 0, s_vw, est, lane);
    au[ib].x += o.x; au[ib].y += o.y;
  }
  __syncthreads();

  // ---- attention 2: user <- friends (W4,W5,W6), no ratings ----
  if (tid < D_DIM) s_vw[tid] = proj[192 + tid];       // v2 = W5@W6
  else             s_vw[tid] = W6[tid - D_DIM];       // w2 (unused: no ratings)
  __syncthreads();
  for (int ib = 0; ib < BPW; ++ib) {
    int b = b0 + wave * BPW + ib;
    float2 o = wave_attention<false>(embed_user, u_friends + (size_t)b * 32,
                                     nullptr, 32, qu[ib],
                                     proj + 128, s_vw, est, lane);
    au[ib].x += o.x; au[ib].y += o.y;
  }
  __syncthreads();

  // ---- attention 3: item <- interacting users (W7,W8,W9) ----
  if (tid < D_DIM) s_vw[tid] = proj[320 + tid];       // v3 = W8@W9
  else             s_vw[tid] = W9[tid - D_DIM];       // w3 = W9
  __syncthreads();
  for (int ib = 0; ib < BPW; ++ib) {
    int b = b0 + wave * BPW + ib;
    float2 o = wave_attention<true>(embed_user, ih_users + (size_t)b * 50,
                                    ih_rat + (size_t)b * 50, 50, qi[ib],
                                    proj + 256, s_vw, est, lane);
    ai[ib].x += o.x; ai[ib].y += o.y;
  }

  // Write concat tile [user_final | item_final] as f16 rows.
  #pragma unroll
  for (int ib = 0; ib < BPW; ++ib) {
    int row = wave * BPW + ib;
    _Float16* cr = s_cat + row * CSTRIDE;
    cr[2 * lane]              = (_Float16)au[ib].x;
    cr[2 * lane + 1]          = (_Float16)au[ib].y;
    cr[D_DIM + 2 * lane]      = (_Float16)ai[ib].x;
    cr[D_DIM + 2 * lane + 1]  = (_Float16)ai[ib].y;
  }
  __syncthreads();

  // ---- fc1 (WMMA f16->f32) + ReLU + fc2, all 4 waves: wave == N-tile ----
  {
    const int mrow = lane & 15;
    const int kb   = (lane < 16) ? 0 : 8;    // A layout: K 0-7/16-23 vs 8-15/24-31
    const int nn   = wave;

    v8f acc = {};
    #pragma unroll
    for (int kk = 0; kk < 4; ++kk) {
      // A fragment from LDS concat tile (two aligned b128 loads).
      const v8h* p0 = (const v8h*)(s_cat + mrow * CSTRIDE + kk * 32 + kb);
      const v8h* p1 = (const v8h*)(s_cat + mrow * CSTRIDE + kk * 32 + kb + 16);
      v8h lo = *p0, hi = *p1;
      v16h af = __builtin_shufflevector(lo, hi, 0, 1, 2, 3, 4, 5, 6, 7,
                                                8, 9, 10, 11, 12, 13, 14, 15);
      // B fragment: pre-packed, coalesced 32B per lane.
      v16h bf = *(const v16h*)(fc1_pack + (size_t)(((nn * 4 + kk) * 32 + lane) << 4));
      acc = __builtin_amdgcn_wmma_f32_16x16x32_f16(
          /*neg_a=*/false, af, /*neg_b=*/false, bf,
          /*c_mod=*/(short)0, acc, /*reuse_a=*/false, /*reuse_b=*/false);
    }

    // ReLU + fc2 partial: this wave covers columns nn*16 .. nn*16+15.
    const int ncol = lane & 15;
    float bias = fc1_b[nn * 16 + ncol];
    float w2   = fc2_w[nn * 16 + ncol];
    float p[8];
    #pragma unroll
    for (int r = 0; r < 8; ++r) {
      float h = fmaxf(acc[r] + bias, 0.0f);
      p[r] = h * w2;
    }
    // Reduce over the 16 N-lanes of each half-wave (rows r / r+8).
    #pragma unroll
    for (int r = 0; r < 8; ++r)
      #pragma unroll
      for (int m = 8; m; m >>= 1) p[r] += __shfl_xor(p[r], m, 16);

    if (lane == 0) {
      #pragma unroll
      for (int r = 0; r < 8; ++r) s_red[wave * 16 + r] = p[r];
    }
    if (lane == 16) {
      #pragma unroll
      for (int r = 0; r < 8; ++r) s_red[wave * 16 + 8 + r] = p[r];
    }
  }
  __syncthreads();

  // Combine the 4 N-tile partials and emit the 16 outputs.
  if (tid < 16) {
    float s = fc2_b[0];
    #pragma unroll
    for (int w = 0; w < WAVES; ++w) s += s_red[w * 16 + tid];
    out[b0 + tid] = s;
  }
}

extern "C" void kernel_launch(void* const* d_in, const int* in_sizes, int n_in,
                              void* d_out, int out_size, void* d_ws, size_t ws_size,
                              hipStream_t stream) {
  const int*   user_ids  = (const int*)  d_in[0];
  const int*   item_ids  = (const int*)  d_in[1];
  const int*   uh_items  = (const int*)  d_in[2];
  const float* uh_rat    = (const float*)d_in[3];
  const int*   u_friends = (const int*)  d_in[4];
  const int*   ih_users  = (const int*)  d_in[5];
  const float* ih_rat    = (const float*)d_in[6];
  const float* emb_user  = (const float*)d_in[7];
  const float* emb_item  = (const float*)d_in[8];
  const float* W1 = (const float*)d_in[9];
  const float* W2 = (const float*)d_in[10];
  const float* W3 = (const float*)d_in[11];
  const float* W4 = (const float*)d_in[12];
  const float* W5 = (const float*)d_in[13];
  const float* W6 = (const float*)d_in[14];
  const float* W7 = (const float*)d_in[15];
  const float* W8 = (const float*)d_in[16];
  const float* W9 = (const float*)d_in[17];
  const float* fc1_w = (const float*)d_in[18];
  const float* fc1_b = (const float*)d_in[19];
  const float* fc2_w = (const float*)d_in[20];
  const float* fc2_b = (const float*)d_in[21];
  float* out = (float*)d_out;

  float*    proj = (float*)d_ws;                            // 384 f32
  _Float16* pack = (_Float16*)((char*)d_ws + PACK_OFF);     // 8192 f16

  const int B = in_sizes[0];

  graphrec_prep<<<1, 384, 0, stream>>>(W1, W2, W3, W4, W5, W6, W7, W8, W9, proj);
  graphrec_pack_b<<<32, 256, 0, stream>>>(fc1_w, pack);

  const int rows_per_block = WAVES * BPW;          // 16
  const int grid = B / rows_per_block;             // B = 16384 -> 1024
  graphrec_main<<<grid, WAVES * 32, 0, stream>>>(
      user_ids, item_ids, uh_items, uh_rat, u_friends, ih_users, ih_rat,
      emb_user, emb_item, W3, W6, W9, pack, fc1_b, fc2_w, fc2_b, proj, out);
}
